// RNNDecoder_13340168421397
// MI455X (gfx1250) — compile-verified, matrix operand order checked by reading, not run on hardware
//
#include <hip/hip_runtime.h>
#include <hip/hip_bf16.h>

// Problem constants (match reference)
#define V_ 32000
#define E_ 512
#define H_ 1024
#define B_ 64
#define T_ 20
#define SOS_IDX 1

typedef __attribute__((ext_vector_type(16))) _Float16 v16h;
typedef __attribute__((ext_vector_type(8)))  _Float16 v8h;
typedef __attribute__((ext_vector_type(8)))  float    v8f;

// ---------------------------------------------------------------------------
// Fragment loader for 16-bit A/B matrices of v_wmma_f32_16x16x32_f16.
// Per ISA layout (CDNA5 ISA §7.12.2), lane (lo = lane&15,
// hi = lane>>4) holds row `lo`, halves 0..7 = K[hi*8 .. hi*8+7],
// halves 8..15 = K[16+hi*8 .. 16+hi*8+7] -> two contiguous 16B loads.
// `p` must already point at rowBase + hi*8 (+ k offset applied by caller).
// ---------------------------------------------------------------------------
__device__ __forceinline__ v16h ld_frag16(const _Float16* p) {
    v8h a = *reinterpret_cast<const v8h*>(p);
    v8h b = *reinterpret_cast<const v8h*>(p + 16);
    v16h r;
#pragma unroll
    for (int i = 0; i < 8; ++i) { r[i] = a[i]; r[i + 8] = b[i]; }
    return r;
}

// ---------------------------------------------------------------------------
// GEMM: D[M,N] = A[M,K] * B[N,K]^T + bias[N]   (A,B f16 row-major, D f32)
// Block = 256 threads = 8 wave32s arranged 2(M) x 4(N).
// Each wave computes a 32x64 tile: 2 A fragments x 4 B fragments ->
// 8 WMMA accumulators per K-step (12 b128 loads / 8 WMMAs).
// Block tile = 64(M) x 256(N). Requires M%64==0, N%256==0, K%32==0.
// ---------------------------------------------------------------------------
__global__ __launch_bounds__(256)
void gemm_f16f32_wmma(const _Float16* __restrict__ A,
                      const _Float16* __restrict__ Bw,
                      const float* __restrict__ bias,
                      float* __restrict__ D,
                      int M, int N, int K) {
    const int lane = threadIdx.x & 31;
    const int wave = threadIdx.x >> 5;
    const int mw = wave >> 2;          // 0..1
    const int nw = wave & 3;           // 0..3
    const int m0 = blockIdx.x * 64 + mw * 32;
    const int n0 = blockIdx.y * 256 + nw * 64;
    const int lo = lane & 15;
    const int hi = lane >> 4;

    const _Float16* arow0 = A + (long)(m0 + lo) * K + hi * 8;
    const _Float16* arow1 = A + (long)(m0 + 16 + lo) * K + hi * 8;
    const _Float16* brow0 = Bw + (long)(n0 + 0 * 16 + lo) * K + hi * 8;
    const _Float16* brow1 = Bw + (long)(n0 + 1 * 16 + lo) * K + hi * 8;
    const _Float16* brow2 = Bw + (long)(n0 + 2 * 16 + lo) * K + hi * 8;
    const _Float16* brow3 = Bw + (long)(n0 + 3 * 16 + lo) * K + hi * 8;

    v8f a00 = {}, a01 = {}, a02 = {}, a03 = {};
    v8f a10 = {}, a11 = {}, a12 = {}, a13 = {};

    for (int k = 0; k < K; k += 32) {
        // prefetch next K-block (emits global_prefetch_b8 on gfx1250)
        __builtin_prefetch(arow0 + k + 64, 0, 1);
        __builtin_prefetch(brow0 + k + 64, 0, 1);
        __builtin_prefetch(brow2 + k + 64, 0, 1);

        v16h fa0 = ld_frag16(arow0 + k);
        v16h fa1 = ld_frag16(arow1 + k);
        v16h b0 = ld_frag16(brow0 + k);
        v16h b1 = ld_frag16(brow1 + k);
        v16h b2 = ld_frag16(brow2 + k);
        v16h b3 = ld_frag16(brow3 + k);
        a00 = __builtin_amdgcn_wmma_f32_16x16x32_f16(false, fa0, false, b0, (short)0, a00, false, false);
        a10 = __builtin_amdgcn_wmma_f32_16x16x32_f16(false, fa1, false, b0, (short)0, a10, false, false);
        a01 = __builtin_amdgcn_wmma_f32_16x16x32_f16(false, fa0, false, b1, (short)0, a01, false, false);
        a11 = __builtin_amdgcn_wmma_f32_16x16x32_f16(false, fa1, false, b1, (short)0, a11, false, false);
        a02 = __builtin_amdgcn_wmma_f32_16x16x32_f16(false, fa0, false, b2, (short)0, a02, false, false);
        a12 = __builtin_amdgcn_wmma_f32_16x16x32_f16(false, fa1, false, b2, (short)0, a12, false, false);
        a03 = __builtin_amdgcn_wmma_f32_16x16x32_f16(false, fa0, false, b3, (short)0, a03, false, false);
        a13 = __builtin_amdgcn_wmma_f32_16x16x32_f16(false, fa1, false, b3, (short)0, a13, false, false);
    }

    // Epilogue. C/D layout: VGPR r -> (M = mBase + r + 8*hi, N = n0 + j*16 + lo)
    v8f accs[2][4] = {{a00, a01, a02, a03}, {a10, a11, a12, a13}};
#pragma unroll
    for (int i = 0; i < 2; ++i) {
        const int mb = m0 + i * 16;
#pragma unroll
        for (int j = 0; j < 4; ++j) {
            const int n = n0 + j * 16 + lo;
            const float bi = bias[n];
#pragma unroll
            for (int r = 0; r < 8; ++r) {
                const int m = mb + r + hi * 8;
                D[(long)m * N + n] = accs[i][j][r] + bi;
            }
        }
    }
}

// ---------------------------------------------------------------------------
// f32 -> f16 conversion (grid-stride)
// ---------------------------------------------------------------------------
__global__ void cvt_f32_f16(const float* __restrict__ src,
                            _Float16* __restrict__ dst, long n) {
    long i = (long)blockIdx.x * blockDim.x + threadIdx.x;
    const long stride = (long)gridDim.x * blockDim.x;
    for (; i < n; i += stride) dst[i] = (_Float16)src[i];
}

// ---------------------------------------------------------------------------
// Embedding gather with SOS prepend: row (b,t) gets emb[t==0 ? SOS : seq[b,t-1]]
// ---------------------------------------------------------------------------
__global__ void gather_embed(const int* __restrict__ seq,
                             const float* __restrict__ emb,
                             _Float16* __restrict__ X) {
    const int row = blockIdx.x;            // 0 .. B_*T_-1
    const int b = row / T_;
    const int t = row % T_;
    const int tok = (t == 0) ? SOS_IDX : seq[b * T_ + (t - 1)];
    const float* e = emb + (long)tok * E_;
    _Float16* x = X + (long)row * E_;
    for (int i = threadIdx.x; i < E_; i += blockDim.x) x[i] = (_Float16)e[i];
}

// ---------------------------------------------------------------------------
// h0 init: copy f32 + make f16 copy (must not mutate d_in)
// ---------------------------------------------------------------------------
__global__ void init_h(const float* __restrict__ h0,
                       float* __restrict__ h32,
                       _Float16* __restrict__ h16) {
    const int i = blockIdx.x * blockDim.x + threadIdx.x;
    if (i < B_ * H_) {
        const float v = h0[i];
        h32[i] = v;
        h16[i] = (_Float16)v;
    }
}

// ---------------------------------------------------------------------------
// GRU gate nonlinearity for timestep t.
// GI [B*T, 3H] precomputed, GH [B, 3H] from current-step GEMM.
// Updates h32/h16 in place (thread-local index) and writes states.
// ---------------------------------------------------------------------------
__global__ void gru_gate(const float* __restrict__ GI,
                         const float* __restrict__ GH,
                         float* __restrict__ h32,
                         _Float16* __restrict__ h16,
                         float* __restrict__ states,       // d_out [B,T,H]
                         _Float16* __restrict__ states16,  // ws f16 copy
                         int t) {
    const int idx = blockIdx.x * blockDim.x + threadIdx.x;
    if (idx >= B_ * H_) return;
    const int b = idx / H_;
    const int h = idx % H_;
    const long gir = ((long)b * T_ + t) * (3 * H_);
    const long ghr = (long)b * (3 * H_);
    const float i_r = GI[gir + h];
    const float i_z = GI[gir + H_ + h];
    const float i_n = GI[gir + 2 * H_ + h];
    const float h_r = GH[ghr + h];
    const float h_z = GH[ghr + H_ + h];
    const float h_n = GH[ghr + 2 * H_ + h];
    const float r = 1.0f / (1.0f + __expf(-(i_r + h_r)));
    const float z = 1.0f / (1.0f + __expf(-(i_z + h_z)));
    const float n = tanhf(i_n + r * h_n);
    const float hp = h32[idx];
    const float hn = (1.0f - z) * n + z * hp;
    h32[idx] = hn;
    h16[idx] = (_Float16)hn;
    const long s = ((long)b * T_ + t) * H_ + h;
    states[s] = hn;
    states16[s] = (_Float16)hn;
}

// ---------------------------------------------------------------------------
// Host-side launcher
// ---------------------------------------------------------------------------
extern "C" void kernel_launch(void* const* d_in, const int* in_sizes, int n_in,
                              void* d_out, int out_size, void* d_ws, size_t ws_size,
                              hipStream_t stream) {
    (void)in_sizes; (void)n_in; (void)out_size; (void)ws_size;

    const int*   seq   = (const int*)  d_in[0];  // [B,T]
    const float* h0    = (const float*)d_in[1];  // [B,H]
    const float* emb   = (const float*)d_in[2];  // [V,E]
    const float* w_ih  = (const float*)d_in[3];  // [3H,E]
    const float* w_hh  = (const float*)d_in[4];  // [3H,H]
    const float* b_ih  = (const float*)d_in[5];  // [3H]
    const float* b_hh  = (const float*)d_in[6];  // [3H]
    const float* w_out = (const float*)d_in[7];  // [V,H]
    const float* b_out = (const float*)d_in[8];  // [V]

    float* logits = (float*)d_out;                                  // [B,T,V]
    float* states = (float*)d_out + (long)B_ * T_ * V_;             // [B,T,H]

    // Workspace carve-up (256B aligned)
    char* w = (char*)d_ws;
    size_t off = 0;
    auto alloc = [&](size_t bytes) -> void* {
        void* p = w + off;
        off = (off + bytes + 255) & ~(size_t)255;
        return p;
    };
    _Float16* WIH16  = (_Float16*)alloc((size_t)3 * H_ * E_ * 2);
    _Float16* WHH16  = (_Float16*)alloc((size_t)3 * H_ * H_ * 2);
    _Float16* WOUT16 = (_Float16*)alloc((size_t)V_ * H_ * 2);
    _Float16* X16    = (_Float16*)alloc((size_t)B_ * T_ * E_ * 2);
    float*    GI     = (float*)   alloc((size_t)B_ * T_ * 3 * H_ * 4);
    float*    GH     = (float*)   alloc((size_t)B_ * 3 * H_ * 4);
    float*    H32    = (float*)   alloc((size_t)B_ * H_ * 4);
    _Float16* H16    = (_Float16*)alloc((size_t)B_ * H_ * 2);
    _Float16* S16    = (_Float16*)alloc((size_t)B_ * T_ * H_ * 2);

    // 1) weight conversions to f16
    cvt_f32_f16<<<2048, 256, 0, stream>>>(w_ih,  WIH16,  (long)3 * H_ * E_);
    cvt_f32_f16<<<2048, 256, 0, stream>>>(w_hh,  WHH16,  (long)3 * H_ * H_);
    cvt_f32_f16<<<8192, 256, 0, stream>>>(w_out, WOUT16, (long)V_ * H_);

    // 2) embedding gather (SOS prepend + shift) -> X16 [B*T, E]
    gather_embed<<<B_ * T_, 256, 0, stream>>>(seq, emb, X16);

    // 3) h0
    init_h<<<(B_ * H_ + 255) / 256, 256, 0, stream>>>(h0, H32, H16);

    // 4) GI = X @ w_ih^T + b_ih : M=1280, N=3072, K=512
    {
        dim3 grid((B_ * T_) / 64, (3 * H_) / 256);
        gemm_f16f32_wmma<<<grid, 256, 0, stream>>>(X16, WIH16, b_ih, GI,
                                                   B_ * T_, 3 * H_, E_);
    }

    // 5) sequential GRU: per step, GH = h @ w_hh^T + b_hh, then gates
    for (int t = 0; t < T_; ++t) {
        dim3 grid(B_ / 64, (3 * H_) / 256);
        gemm_f16f32_wmma<<<grid, 256, 0, stream>>>(H16, WHH16, b_hh, GH,
                                                   B_, 3 * H_, H_);
        gru_gate<<<(B_ * H_ + 255) / 256, 256, 0, stream>>>(GI, GH, H32, H16,
                                                            states, S16, t);
    }

    // 6) logits = states @ w_out^T + b_out : M=1280, N=32000, K=1024
    {
        dim3 grid((B_ * T_) / 64, V_ / 256);
        gemm_f16f32_wmma<<<grid, 256, 0, stream>>>(S16, WOUT16, b_out, logits,
                                                   B_ * T_, V_, H_);
    }
}
